// GenomicGATv2_14585708937429
// MI455X (gfx1250) — compile-verified, matrix operand order checked by reading, not run on hardware
//
#include <hip/hip_runtime.h>
#include <hip/hip_bf16.h>

#define NNODES 50000
#define NEDGES 800000
#define NEG_SLOPE 0.2f

typedef __attribute__((ext_vector_type(16))) __bf16 v16bf;
typedef __attribute__((ext_vector_type(8)))  float  v8f;

__device__ inline unsigned short f2bf(float f) {
  unsigned u = __float_as_uint(f);
  u = (u + 0x7fffu + ((u >> 16) & 1u)) >> 16;   // round-to-nearest-even
  return (unsigned short)u;
}
// monotone unsigned encoding of float for atomicMax-based segment max
__device__ inline unsigned f32_to_ord(float f) {
  unsigned u = __float_as_uint(f);
  return (u & 0x80000000u) ? ~u : (u | 0x80000000u);
}
__device__ inline float ord_to_f32(unsigned u) {
  return (u & 0x80000000u) ? __uint_as_float(u & 0x7fffffffu) : __uint_as_float(~u);
}

union FragAB { v16bf v; unsigned u[8]; };

// 16x32 bf16 fragment load (A layout; B uses same layout with pre-transposed W).
// lane&15 = row/col, lane>>4 selects K-half groups; VGPR j packs K pairs.
__device__ inline v16bf load_frag(const unsigned short* tile, int ld) {
  int lane = threadIdx.x & 31;
  int r = lane & 15;
  int kb = (lane >> 4) << 3;           // 0 or 8
  const unsigned short* row = tile + (size_t)r * ld;
  FragAB f;
#pragma unroll
  for (int j = 0; j < 4; ++j)
    f.u[j] = *(const unsigned*)(row + kb + 2 * j);
#pragma unroll
  for (int j = 0; j < 4; ++j)
    f.u[4 + j] = *(const unsigned*)(row + 16 + kb + 2 * j);
  return f.v;
}

// C = A[M,K](bf16) * BT[N,K](bf16)^T + bias[N], one 16x16 tile per wave.
__global__ void __launch_bounds__(256)
gemm_bf16_wmma(const unsigned short* __restrict__ A,
               const unsigned short* __restrict__ BT,
               const float* __restrict__ bias, float* __restrict__ C,
               int M, int N, int K) {
  int wave = blockIdx.x * (blockDim.x >> 5) + (threadIdx.x >> 5);
  int tilesN = N >> 4;
  int numTiles = (M >> 4) * tilesN;
  if (wave >= numTiles) return;                 // wave-uniform branch (EXEC stays full)
  int tm = wave / tilesN;
  int tn = wave - tm * tilesN;
  const unsigned short* a = A + (size_t)tm * 16 * K;
  const unsigned short* b = BT + (size_t)tn * 16 * K;
  v8f acc = {0.f, 0.f, 0.f, 0.f, 0.f, 0.f, 0.f, 0.f};
  for (int k = 0; k < K; k += 32) {
    v16bf af = load_frag(a + k, K);
    v16bf bf = load_frag(b + k, K);
    acc = __builtin_amdgcn_wmma_f32_16x16x32_bf16(false, af, false, bf,
                                                  (short)0, acc, false, false);
  }
  int lane = threadIdx.x & 31;
  int n = lane & 15;
  int mb = (lane >> 4) << 3;
  float bv = bias[tn * 16 + n];
  float* cp = C + (size_t)(tm * 16 + mb) * N + tn * 16 + n;
#pragma unroll
  for (int r = 0; r < 8; ++r)
    cp[(size_t)r * N] = acc[r] + bv;
}

// ---------------- packing kernels ----------------
__global__ void pack_x_bf16(const float* __restrict__ x,
                            unsigned short* __restrict__ xb, int n) {
  int t = blockIdx.x * blockDim.x + threadIdx.x;
  if (t < n) xb[t] = f2bf(x[t]);
}

// wT1[512,128] = concat(W_src1,W_dst1)^T ; bias1[512]
__global__ void pack_w1(const float* __restrict__ Ws, const float* __restrict__ Wd,
                        const float* __restrict__ bs, const float* __restrict__ bd,
                        unsigned short* __restrict__ wT, float* __restrict__ bias) {
  int t = blockIdx.x * blockDim.x + threadIdx.x;
  if (t >= 512 * 128) return;
  int n = t >> 7, k = t & 127;
  float w = (n < 256) ? Ws[(size_t)k * 256 + n] : Wd[(size_t)k * 256 + (n - 256)];
  wT[(size_t)n * 128 + k] = f2bf(w);
  if (t < 512) bias[t] = (t < 256) ? bs[t] : bd[t - 256];
}

// wT2[32,256] = concat(W_src2,W_dst2)^T ; bias2[32]
__global__ void pack_w2(const float* __restrict__ Ws, const float* __restrict__ Wd,
                        const float* __restrict__ bs, const float* __restrict__ bd,
                        unsigned short* __restrict__ wT, float* __restrict__ bias) {
  int t = blockIdx.x * blockDim.x + threadIdx.x;
  if (t >= 32 * 256) return;
  int n = t >> 8, k = t & 255;
  float w = (n < 16) ? Ws[(size_t)k * 16 + n] : Wd[(size_t)k * 16 + (n - 16)];
  wT[(size_t)n * 256 + k] = f2bf(w);
  if (t < 32) bias[t] = (t < 16) ? bs[t] : bd[t - 16];
}

// ---------------- layer-1 edge/node kernels ----------------
// thread = (edge, head): score + per-dst atomic max (ordered-int)
__global__ void edge_score_max_l1(const int* __restrict__ src, const int* __restrict__ dst,
                                  const float* __restrict__ hsd, const float* __restrict__ attn,
                                  float* __restrict__ e_out, unsigned* __restrict__ m_ord) {
  int t = blockIdx.x * blockDim.x + threadIdx.x;
  if (t >= NEDGES * 4) return;
  int e = t >> 2, h = t & 3;
  int s = src[e], d = dst[e];
  const float* hs = hsd + (size_t)s * 512 + h * 64;        // hs block: cols 0..255
  const float* hd = hsd + (size_t)d * 512 + 256 + h * 64;  // hd block: cols 256..511
  const float* a = attn + h * 64;
  float acc = 0.f;
#pragma unroll 8
  for (int f = 0; f < 64; ++f) {
    float v = hs[f] + hd[f];
    v = v > 0.f ? v : NEG_SLOPE * v;
    acc += a[f] * v;
  }
  e_out[t] = acc;
  atomicMax(&m_ord[(size_t)d * 4 + h], f32_to_ord(acc));
}

// thread = (edge, head): ex = exp(e-m); denom += ex; acc[dst] += ex*hs[src]
__global__ void edge_soft_acc_l1(const int* __restrict__ src, const int* __restrict__ dst,
                                 const float* __restrict__ hsd, const float* __restrict__ e_in,
                                 const unsigned* __restrict__ m_ord,
                                 float* __restrict__ s_sum, float* __restrict__ acc_out) {
  int t = blockIdx.x * blockDim.x + threadIdx.x;
  if (t >= NEDGES * 4) return;
  int e = t >> 2, h = t & 3;
  int s = src[e], d = dst[e];
  float ex = __expf(e_in[t] - ord_to_f32(m_ord[(size_t)d * 4 + h]));
  atomicAdd(&s_sum[(size_t)d * 4 + h], ex);
  const float* hs = hsd + (size_t)s * 512 + h * 64;
  float* ao = acc_out + (size_t)d * 256 + h * 64;
#pragma unroll 8
  for (int f = 0; f < 64; ++f)
    atomicAdd(&ao[f], ex * hs[f]);
}

// normalize + ReLU + repack bf16 for layer-2 GEMM
__global__ void node_finish_l1(const float* __restrict__ acc, const float* __restrict__ s_sum,
                               unsigned short* __restrict__ h1b) {
  int t = blockIdx.x * blockDim.x + threadIdx.x;
  if (t >= NNODES * 256) return;
  int node = t >> 8, h = (t & 255) >> 6;
  float s = s_sum[(size_t)node * 4 + h];
  float v = (s > 0.f) ? acc[t] / s : 0.f;
  h1b[t] = f2bf(v > 0.f ? v : 0.f);
}

// ---------------- layer-2 edge/node kernels (H=1, F=16) ----------------
__global__ void edge_score_max_l2(const int* __restrict__ src, const int* __restrict__ dst,
                                  const float* __restrict__ hsd2, const float* __restrict__ attn2,
                                  float* __restrict__ e_out, unsigned* __restrict__ m_ord) {
  int e = blockIdx.x * blockDim.x + threadIdx.x;
  if (e >= NEDGES) return;
  int s = src[e], d = dst[e];
  const float* hs = hsd2 + (size_t)s * 32;        // cols 0..15
  const float* hd = hsd2 + (size_t)d * 32 + 16;   // cols 16..31
  float acc = 0.f;
#pragma unroll
  for (int f = 0; f < 16; ++f) {
    float v = hs[f] + hd[f];
    v = v > 0.f ? v : NEG_SLOPE * v;
    acc += attn2[f] * v;
  }
  e_out[e] = acc;
  atomicMax(&m_ord[d], f32_to_ord(acc));
}

__global__ void edge_soft_acc_l2(const int* __restrict__ src, const int* __restrict__ dst,
                                 const float* __restrict__ hsd2, const float* __restrict__ e_in,
                                 const unsigned* __restrict__ m_ord,
                                 float* __restrict__ s_sum, float* __restrict__ acc_out) {
  int e = blockIdx.x * blockDim.x + threadIdx.x;
  if (e >= NEDGES) return;
  int s = src[e], d = dst[e];
  float ex = __expf(e_in[e] - ord_to_f32(m_ord[d]));
  atomicAdd(&s_sum[d], ex);
  const float* hs = hsd2 + (size_t)s * 32;
  float* ao = acc_out + (size_t)d * 16;
#pragma unroll
  for (int f = 0; f < 16; ++f)
    atomicAdd(&ao[f], ex * hs[f]);
}

__global__ void node_finish_l2(const float* __restrict__ acc, const float* __restrict__ s_sum,
                               float* __restrict__ out) {
  int t = blockIdx.x * blockDim.x + threadIdx.x;
  if (t >= NNODES * 16) return;
  int node = t >> 4;
  float s = s_sum[node];
  out[t] = (s > 0.f) ? acc[t] / s : 0.f;
}

static inline int cdiv(long long a, int b) { return (int)((a + b - 1) / b); }

extern "C" void kernel_launch(void* const* d_in, const int* in_sizes, int n_in,
                              void* d_out, int out_size, void* d_ws, size_t ws_size,
                              hipStream_t stream) {
  const float* x     = (const float*)d_in[0];
  const int*   src   = (const int*)  d_in[1];
  const int*   dst   = (const int*)  d_in[2];
  const float* Ws1   = (const float*)d_in[3];
  const float* bs1   = (const float*)d_in[4];
  const float* Wd1   = (const float*)d_in[5];
  const float* bd1   = (const float*)d_in[6];
  const float* attn1 = (const float*)d_in[7];
  const float* Ws2   = (const float*)d_in[8];
  const float* bs2   = (const float*)d_in[9];
  const float* Wd2   = (const float*)d_in[10];
  const float* bd2   = (const float*)d_in[11];
  const float* attn2 = (const float*)d_in[12];
  float* out = (float*)d_out;

  // ---- workspace layout (bytes; all chunks multiple of 256) ----
  unsigned char* p = (unsigned char*)d_ws;
  unsigned short* xb   = (unsigned short*)p; p += (size_t)NNODES * 128 * 2;   // 12.8 MB
  unsigned short* wT1  = (unsigned short*)p; p += (size_t)512 * 128 * 2;      // 128 KB
  float*          b1   = (float*)p;          p += 512 * 4;
  float*          hsd1 = (float*)p;          p += (size_t)NNODES * 512 * 4;   // 102.4 MB (reused as hsd2)
  float*          e1   = (float*)p;          p += (size_t)NEDGES * 4 * 4;     // 12.8 MB (reused as e2)
  unsigned*       m1   = (unsigned*)p;       p += (size_t)NNODES * 4 * 4;     // (reused as m2)
  float*          s1   = (float*)p;          p += (size_t)NNODES * 4 * 4;     // (reused as s2)
  float*          acc1 = (float*)p;          p += (size_t)NNODES * 256 * 4;   // 51.2 MB (reused as acc2)
  unsigned short* h1b  = (unsigned short*)p; p += (size_t)NNODES * 256 * 2;   // 25.6 MB
  unsigned short* wT2  = (unsigned short*)p; p += (size_t)32 * 256 * 2;
  float*          b2   = (float*)p;          p += 256;

  // contiguous m1|s1|acc1 zero region
  size_t zero_bytes = (size_t)NNODES * 4 * 4 * 2 + (size_t)NNODES * 256 * 4;

  // ---- layer 1 ----
  hipMemsetAsync(m1, 0, zero_bytes, stream);

  pack_x_bf16<<<cdiv((long long)NNODES * 128, 256), 256, 0, stream>>>(x, xb, NNODES * 128);
  pack_w1<<<cdiv(512 * 128, 256), 256, 0, stream>>>(Ws1, Wd1, bs1, bd1, wT1, b1);

  // hs||hd : [N,512] = xb[N,128] @ wT1^T, bf16 WMMA
  {
    int tiles = (NNODES / 16) * (512 / 16);          // 100000
    gemm_bf16_wmma<<<cdiv(tiles, 8), 256, 0, stream>>>(xb, wT1, b1, hsd1, NNODES, 512, 128);
  }

  edge_score_max_l1<<<cdiv((long long)NEDGES * 4, 256), 256, 0, stream>>>(src, dst, hsd1, attn1, e1, m1);
  edge_soft_acc_l1<<<cdiv((long long)NEDGES * 4, 256), 256, 0, stream>>>(src, dst, hsd1, e1, m1, s1, acc1);
  node_finish_l1<<<cdiv((long long)NNODES * 256, 256), 256, 0, stream>>>(acc1, s1, h1b);

  // ---- layer 2 (reuse hsd1->hsd2, e1->e2, m1->m2, s1->s2, acc1->acc2) ----
  hipMemsetAsync(m1, 0, zero_bytes, stream);
  pack_w2<<<cdiv(32 * 256, 256), 256, 0, stream>>>(Ws2, Wd2, bs2, bd2, wT2, b2);

  float* hsd2 = hsd1;
  {
    int tiles = (NNODES / 16) * (32 / 16);           // 6250
    gemm_bf16_wmma<<<cdiv(tiles, 8), 256, 0, stream>>>(h1b, wT2, b2, hsd2, NNODES, 32, 256);
  }

  edge_score_max_l2<<<cdiv(NEDGES, 256), 256, 0, stream>>>(src, dst, hsd2, attn2, e1, m1);
  edge_soft_acc_l2<<<cdiv(NEDGES, 256), 256, 0, stream>>>(src, dst, hsd2, e1, m1, s1, acc1);
  node_finish_l2<<<cdiv((long long)NNODES * 16, 256), 256, 0, stream>>>(acc1, s1, out);

  (void)in_sizes; (void)n_in; (void)out_size; (void)ws_size;
}
